// SparseConv2D_70222715290210
// MI455X (gfx1250) — compile-verified, matrix operand order checked by reading, not run on hardware
//
#include <hip/hip_runtime.h>
#include <hip/hip_bf16.h>

// SparseConv2D for MI455X (gfx1250): bf16 WMMA implicit-GEMM per 16x16 block,
// with per-block mask gating. ~120 GFLOP effective vs ~18us HBM floor
// (410MB @ 23.3TB/s) -> compute/memory balanced; bf16 WMMA path mandatory.
// Output stores are non-temporal so the streamed 205MB output doesn't evict
// the f32 input (205MB, reused by 4 F-groups per block) from the 192MB L2.

typedef __bf16 bf16;
typedef __attribute__((ext_vector_type(16))) __bf16 v16bf;
typedef __attribute__((ext_vector_type(8)))  __bf16 v8bf;
typedef __attribute__((ext_vector_type(8)))  float  v8f;

#define N_    4
#define H_    224
#define C_    256
#define F_    256
#define BS_   16
#define OB_   14
#define BC_   16
#define NPIX  196          // 14*14 output pixels per block
#define MT_   13           // ceil(196/16) M-tiles
#define NBLK  (N_*BC_*BC_) // 1024

// ---------------------------------------------------------------------------
// Kernel 1: per-block active flag = (mean of zero-padded 16x16 mask window) > 0.5
// ---------------------------------------------------------------------------
__global__ void flags_kernel(const float* __restrict__ mask, int* __restrict__ flags) {
    int b = blockIdx.x * blockDim.x + threadIdx.x;
    if (b >= NBLK) return;
    int n = b >> 8, rem = b & 255, by = rem >> 4, bx = rem & 15;
    float s = 0.f;
    for (int y = 0; y < BS_; ++y) {
        int gy = by * OB_ + y - 1;
        if ((unsigned)gy >= (unsigned)H_) continue;
        const float* row = mask + ((size_t)n * H_ + gy) * H_;
        for (int x = 0; x < BS_; ++x) {
            int gx = bx * OB_ + x - 1;
            if ((unsigned)gx < (unsigned)H_) s += row[gx];
        }
    }
    flags[b] = (s * (1.0f / 256.0f) > 0.5f) ? 1 : 0;
}

// ---------------------------------------------------------------------------
// Kernel 2: repack weights (3,3,C,F) f32 -> bf16 wT[cc][kk][f][k], k=0..31
// contiguous so a lane's B fragment is two aligned 16B chunks (K-halves).
// ---------------------------------------------------------------------------
__global__ void wtrans_kernel(const float* __restrict__ w, bf16* __restrict__ wT) {
    int idx = blockIdx.x * blockDim.x + threadIdx.x;
    if (idx >= 8 * 9 * F_ * 32) return;            // 589824
    int k  = idx & 31;
    int f  = (idx >> 5) & 255;
    int kk = (idx >> 13) % 9;                      // ky*3+kx
    int cc = idx / (9 * 8192);
    int c  = cc * 32 + k;
    wT[idx] = (bf16)w[((size_t)kk * C_ + c) * F_ + f];
}

// ---------------------------------------------------------------------------
// Kernel 3: main conv. One workgroup = (block b, 64-channel F-group).
// 8 waves; wave w owns M-tiles {w, w+8} x 4 N-tiles (16 ch each).
// K loop: 8 channel-chunks x 9 taps = 72 steps of v_wmma_f32_16x16x32_bf16.
// ---------------------------------------------------------------------------
__global__ void __launch_bounds__(256) conv_kernel(
    const float* __restrict__ in, const float* __restrict__ bias,
    const bf16* __restrict__ wT, const int* __restrict__ flags,
    float* __restrict__ out) {

    __shared__ bf16 sIn[BS_][BS_][32];   // 16 KB: input tile, one 32-ch chunk

    const int b     = blockIdx.x;
    const int fbase = blockIdx.y * 64;
    const int n = b >> 8, rem = b & 255, by = rem >> 4, bx = rem & 15;
    const int tid = threadIdx.x;

    if (!flags[b]) {  // inactive block: zero our 196x64 slice (block-uniform)
        for (int i = tid; i < NPIX * 64; i += 256) {
            int p = i >> 6, f = fbase + (i & 63);
            int oy = p / OB_, ox = p % OB_;
            __builtin_nontemporal_store(0.f,
                &out[(((size_t)n * H_ + by * OB_ + oy) * H_ + bx * OB_ + ox) * F_ + f]);
        }
        return;
    }

    const int wave  = tid >> 5;
    const int lane  = tid & 31;
    const int lhalf = lane >> 4;       // which K-half this lane holds (A/B layout)
    const int l15   = lane & 15;       // M for A-frag, N for B-frag / C-tile
    const int klo   = lhalf * 8;       // lane's first K chunk: lanes<16 -> K0..7, else K8..15

    v8f acc[2][4] = {};                // 2 M-tiles x 4 N-tiles, f32 accum
    const int mts[2] = { wave, wave + 8 };

    for (int cc = 0; cc < 8; ++cc) {
        __syncthreads();
        // Stage input chunk: 16x16 positions x 32 ch, f32 -> bf16. 2048 float4s.
        for (int j = tid; j < 2048; j += 256) {
            int yx = j >> 3, cq = j & 7;
            int y = yx >> 4, x = yx & 15;
            int gy = by * OB_ + y - 1, gx = bx * OB_ + x - 1;
            float4 v = make_float4(0.f, 0.f, 0.f, 0.f);
            if ((unsigned)gy < (unsigned)H_ && (unsigned)gx < (unsigned)H_)
                v = *(const float4*)(in + (((size_t)n * H_ + gy) * H_ + gx) * C_ + cc * 32 + cq * 4);
            bf16* d = &sIn[y][x][cq * 4];
            d[0] = (bf16)v.x; d[1] = (bf16)v.y; d[2] = (bf16)v.z; d[3] = (bf16)v.w;
        }
        __syncthreads();

        for (int kk = 0; kk < 9; ++kk) {
            const int ky = kk / 3, kx = kk % 3;

            // B fragments: weights stream from L2 (whole repacked tensor = 1.2MB).
            // Lane layout (16-bit B 32x16): N = lane%16; lanes<16 hold K 0..7 &
            // 16..23, lanes>=16 hold K 8..15 & 24..31 -> two 16B chunks.
            const bf16* wbase = wT + ((size_t)(cc * 9 + kk) * F_) * 32;
            // Prefetch next tap's B panel into cache hierarchy.
            __builtin_prefetch(wbase + (size_t)F_ * 32, 0, 1);
            v16bf bfrag[4];
            for (int nt = 0; nt < 4; ++nt) {
                const bf16* col = wbase + (size_t)(fbase + nt * 16 + l15) * 32;
                v8bf lo = *(const v8bf*)(col + klo);
                v8bf hi = *(const v8bf*)(col + klo + 16);
                bfrag[nt] = __builtin_shufflevector(lo, hi,
                    0,1,2,3,4,5,6,7,8,9,10,11,12,13,14,15);
            }

            for (int mi = 0; mi < 2; ++mi) {
                int mt = mts[mi];
                if (mt >= MT_) break;   // wave-uniform: EXEC stays all-1s inside
                // A fragment (16x32 bf16): row M = lane%16 -> pixel p, im2col
                // column = input (py+ky, px+kx, cc*32 + K). Two 16B LDS chunks.
                int p = mt * 16 + l15; if (p >= NPIX) p = NPIX - 1; // pad rows: dup, discarded at store
                int py = p / OB_ + ky, px = p % OB_ + kx;
                const bf16* arow = &sIn[py][px][0];
                v8bf lo = *(const v8bf*)(arow + klo);
                v8bf hi = *(const v8bf*)(arow + klo + 16);
                v16bf afrag = __builtin_shufflevector(lo, hi,
                    0,1,2,3,4,5,6,7,8,9,10,11,12,13,14,15);

                for (int nt = 0; nt < 4; ++nt)
                    acc[mi][nt] = __builtin_amdgcn_wmma_f32_16x16x32_bf16(
                        false, afrag, false, bfrag[nt], (short)0, acc[mi][nt],
                        false, false);
            }
        }
    }

    // Epilogue: bias + ReLU, non-temporal f32 stores.
    // C/D layout: N = lane%16, M = r + 8*(lane/16).
    float bv[4];
    for (int nt = 0; nt < 4; ++nt) bv[nt] = bias[fbase + nt * 16 + l15];

    for (int mi = 0; mi < 2; ++mi) {
        int mt = mts[mi];
        if (mt >= MT_) break;
        for (int nt = 0; nt < 4; ++nt) {
            int f = fbase + nt * 16 + l15;
            for (int r = 0; r < 8; ++r) {
                int p = mt * 16 + r + lhalf * 8;
                if (p >= NPIX) continue;
                int oy = p / OB_, ox = p % OB_;
                float v = fmaxf(acc[mi][nt][r] + bv[nt], 0.f);
                __builtin_nontemporal_store(v,
                    &out[(((size_t)n * H_ + by * OB_ + oy) * H_ + bx * OB_ + ox) * F_ + f]);
            }
        }
    }
}

// ---------------------------------------------------------------------------
extern "C" void kernel_launch(void* const* d_in, const int* in_sizes, int n_in,
                              void* d_out, int out_size, void* d_ws, size_t ws_size,
                              hipStream_t stream) {
    const float* in   = (const float*)d_in[0];
    const float* mask = (const float*)d_in[1];
    const float* kern = (const float*)d_in[2];
    const float* bias = (const float*)d_in[3];
    float* out = (float*)d_out;

    int*  flags = (int*)d_ws;                       // 1024 ints
    bf16* wT    = (bf16*)((char*)d_ws + 4096);      // 589824 bf16 = 1.18 MB

    flags_kernel<<<(NBLK + 255) / 256, 256, 0, stream>>>(mask, flags);
    wtrans_kernel<<<(8 * 9 * F_ * 32 + 255) / 256, 256, 0, stream>>>(kern, wT);

    dim3 grid(NBLK, 4);   // 1024 blocks x 4 F-groups of 64 channels
    conv_kernel<<<grid, 256, 0, stream>>>(in, bias, wT, flags, out);
}
